// Self_attention_65386582114919
// MI455X (gfx1250) — matrix operand order repaired and checked by reading, not compiled, not run on hardware
//
#include <hip/hip_runtime.h>
#include <hip/hip_bf16.h>

#define LSEQ 1024
#define EDIM 256
#define NH   8
#define HD   32
#define LE   (LSEQ * EDIM)
#define EE   (EDIM * EDIM)

typedef __attribute__((ext_vector_type(16))) _Float16 v16h;
typedef __attribute__((ext_vector_type(8)))  _Float16 v8h;
typedef __attribute__((ext_vector_type(8)))  float    v8f;

// ---------------------------------------------------------------------------
// A-fragment (16x32 f16, MxK): lane l holds row (l&15); K-chunks
// {8h..8h+7} and {16+8h..16+8h+7} with h = l>>4 (ISA 7.12.2, 16-bit A 16x32).
// Source must be row-major with K contiguous. Two 16B loads per lane.
// ---------------------------------------------------------------------------
__device__ __forceinline__ v16h load_fragA(const _Float16* __restrict__ base,
                                           int ld, int row_base, int col_base) {
  const int lane = threadIdx.x & 31;
  const int m = lane & 15, half = lane >> 4;
  const _Float16* p = base + (size_t)(row_base + m) * ld + col_base + 8 * half;
  const v8h c0 = *(const v8h*)(p);
  const v8h c1 = *(const v8h*)(p + 16);
  v16h out;
#pragma unroll
  for (int i = 0; i < 8; ++i) { out[i] = c0[i]; out[i + 8] = c1[i]; }
  return out;
}

// ---------------------------------------------------------------------------
// B-fragment (32x16 f16, KxN): lane l holds column (l&15); K-chunks
// {8h..8h+7} and {16+8h..16+8h+7} with h = l>>4 (same order as A).
// Source: row n with K contiguous.
// ---------------------------------------------------------------------------
__device__ __forceinline__ v16h load_fragB(const _Float16* __restrict__ base,
                                           int ld, int n_base, int k_base) {
  const int lane = threadIdx.x & 31;
  const int n = lane & 15, half = lane >> 4;
  const _Float16* p = base + (size_t)(n_base + n) * ld + k_base + 8 * half;
  const v8h c0 = *(const v8h*)(p);
  const v8h c1 = *(const v8h*)(p + 16);
  v16h out;
#pragma unroll
  for (int i = 0; i < 8; ++i) { out[i] = c0[i]; out[i + 8] = c1[i]; }
  return out;
}

// ---------------------------------------------------------------------------
// f32 -> f16 staging of all matrices that feed WMMA.
// ---------------------------------------------------------------------------
__global__ __launch_bounds__(256) void convert_kernel(
    const float* __restrict__ V, const float* __restrict__ K,
    const float* __restrict__ Q, const float* __restrict__ Wq,
    const float* __restrict__ Wk, const float* __restrict__ Wv,
    const float* __restrict__ Wo,
    _Float16* __restrict__ Vh, _Float16* __restrict__ Kh,
    _Float16* __restrict__ Qh, _Float16* __restrict__ Wqh,
    _Float16* __restrict__ Wkh, _Float16* __restrict__ Wvh,
    _Float16* __restrict__ Woh) {
  const int i = blockIdx.x * 256 + threadIdx.x;
  if (i < LE) {
    Vh[i] = (_Float16)V[i];
    Kh[i] = (_Float16)K[i];
    Qh[i] = (_Float16)Q[i];
  }
  if (i < EE) {
    Wqh[i] = (_Float16)Wq[i];
    Wkh[i] = (_Float16)Wk[i];
    Wvh[i] = (_Float16)Wv[i];
    Woh[i] = (_Float16)Wo[i];
  }
}

// ---------------------------------------------------------------------------
// Generic WMMA GEMM: C = A(MxK, row-major) * B(NxK, row-major)^T.
// One wave owns a 16x16 tile, K-loop in steps of 32 (8 WMMAs for K=256).
// Optional outputs: f16 row-major, f16 transposed (for vT), f32 (+bias).
// ---------------------------------------------------------------------------
__global__ __launch_bounds__(128) void wmma_gemm(
    const _Float16* __restrict__ A, const _Float16* __restrict__ B,
    int M, int N, int K,
    _Float16* __restrict__ Crow, _Float16* __restrict__ Ctr,
    float* __restrict__ Cf, const float* __restrict__ bias) {
  const int wave = (int)((blockIdx.x * blockDim.x + threadIdx.x) >> 5);
  const int lane = threadIdx.x & 31;
  const int ntiles = N >> 4;
  const int mt = wave / ntiles;
  const int nt = wave - mt * ntiles;
  if (mt >= (M >> 4)) return;          // wave-uniform: EXEC stays all-ones

  v8f acc = {};
  for (int k = 0; k < K; k += 32) {
    const v16h a = load_fragA(A, K, mt * 16, k);
    const v16h b = load_fragB(B, K, nt * 16, k);
    acc = __builtin_amdgcn_wmma_f32_16x16x32_f16(false, a, false, b,
                                                 (short)0, acc, false, false);
  }

  const int n = nt * 16 + (lane & 15);
  const int mbase = mt * 16 + 8 * (lane >> 4);
  const float bv = bias ? bias[n] : 0.0f;
#pragma unroll
  for (int r = 0; r < 8; ++r) {
    const int m = mbase + r;
    const float v = acc[r] + bv;
    if (Crow) Crow[(size_t)m * N + n] = (_Float16)v;
    if (Ctr)  Ctr[(size_t)n * M + m] = (_Float16)v;   // transposed store (vT)
    if (Cf)   Cf[(size_t)m * N + n] = v;
  }
}

// ---------------------------------------------------------------------------
// coeff[h, m] = sum_d q[m, h, d] * Wr[h*D + d]
// ---------------------------------------------------------------------------
__global__ __launch_bounds__(256) void coeff_kernel(
    const _Float16* __restrict__ qh, const float* __restrict__ Wr,
    float* __restrict__ coeff) {
  const int t = blockIdx.x * 256 + threadIdx.x;   // 0 .. NH*LSEQ-1
  const int h = t >> 10;
  const int m = t & (LSEQ - 1);
  float s = 0.0f;
#pragma unroll
  for (int d = 0; d < HD; ++d)
    s += (float)qh[(size_t)m * EDIM + h * HD + d] * Wr[h * HD + d];
  coeff[t] = s;
}

// ---------------------------------------------------------------------------
// Flash attention: one wave per (head, 16-row q tile); K/V streamed in tiles
// of 32 rows. S tiles via WMMA; bias coeff[h,q]*log(|pq-pk|+1) added in-reg;
// online softmax with cross-lane (row) reductions; P transposed C->A frag
// through a 1KB per-wave LDS tile; P*V via two WMMAs per step.
// ---------------------------------------------------------------------------
__global__ __launch_bounds__(128) void attn_kernel(
    const _Float16* __restrict__ qh, const _Float16* __restrict__ kh,
    const _Float16* __restrict__ vT, const float* __restrict__ position,
    const float* __restrict__ coeff, _Float16* __restrict__ oh) {
  __shared__ __align__(16) _Float16 ldsP[4][16 * 32];
  const int wib  = threadIdx.x >> 5;
  const int lane = threadIdx.x & 31;
  const int unit = blockIdx.x * 4 + wib;            // NH * (LSEQ/16) = 512
  if (unit >= NH * (LSEQ / 16)) return;             // wave-uniform
  const int h     = unit >> 6;
  const int qt    = unit & 63;
  const int qbase = qt * 16;
  const int dbase = h * HD;
  const int n = lane & 15, half = lane >> 4;

  const v16h aq = load_fragA(qh, EDIM, qbase, dbase);   // whole D=32 in one frag

  float posq[8], cf[8], mi[8], li[8];
#pragma unroll
  for (int r = 0; r < 8; ++r) {
    const int m = qbase + r + 8 * half;
    posq[r] = position[m];
    cf[r]   = coeff[h * LSEQ + m];
    mi[r]   = -3.0e38f;
    li[r]   = 0.0f;
  }
  v8f acc0 = {}, acc1 = {};
  const v8f zero = {};
  const float scale = 0.0625f;                 // 1/sqrt(256)
  _Float16* Pl = ldsP[wib];

  for (int kb = 0; kb < LSEQ; kb += 32) {
    // ---- S = q . k^T  (two 16x16 tiles covering 32 keys) ----
    const v16h bk0 = load_fragB(kh, EDIM, kb,      dbase);
    const v16h bk1 = load_fragB(kh, EDIM, kb + 16, dbase);
    v8f s0 = __builtin_amdgcn_wmma_f32_16x16x32_f16(false, aq, false, bk0,
                                                    (short)0, zero, false, false);
    v8f s1 = __builtin_amdgcn_wmma_f32_16x16x32_f16(false, aq, false, bk1,
                                                    (short)0, zero, false, false);
    const float pk0 = position[kb + n];
    const float pk1 = position[kb + 16 + n];

    // ---- bias + online softmax (rows live across 16 lanes of a half) ----
#pragma unroll
    for (int r = 0; r < 8; ++r) {
      float e0 = (s0[r] + cf[r] * __logf(fabsf(posq[r] - pk0) + 1.0f)) * scale;
      float e1 = (s1[r] + cf[r] * __logf(fabsf(posq[r] - pk1) + 1.0f)) * scale;
      float t = fmaxf(e0, e1);
#pragma unroll
      for (int off = 1; off < 16; off <<= 1) t = fmaxf(t, __shfl_xor(t, off, 32));
      const float mn    = fmaxf(mi[r], t);
      const float alpha = __expf(mi[r] - mn);
      const float p0 = __expf(e0 - mn);
      const float p1 = __expf(e1 - mn);
      float rs = p0 + p1;
#pragma unroll
      for (int off = 1; off < 16; off <<= 1) rs += __shfl_xor(rs, off, 32);
      li[r] = li[r] * alpha + rs;
      mi[r] = mn;
      acc0[r] *= alpha;
      acc1[r] *= alpha;
      // P into LDS as row-major 16x32 f16 (C-frag coords -> memory)
      const int prow = (r + 8 * half) * 32;
      Pl[prow + n]      = (_Float16)p0;
      Pl[prow + 16 + n] = (_Float16)p1;
    }
    asm volatile("s_wait_dscnt 0" ::: "memory");

    // ---- re-load P as an A fragment (16x32) from LDS ----
    v16h ap;
    {
      const _Float16* p = Pl + (lane & 15) * 32 + 8 * half;
      const v8h c0 = *(const v8h*)(p);
      const v8h c1 = *(const v8h*)(p + 16);
#pragma unroll
      for (int i = 0; i < 8; ++i) { ap[i] = c0[i]; ap[i + 8] = c1[i]; }
    }

    // ---- acc += P * V  (B frags from transposed V: K contiguous) ----
    const v16h bv0 = load_fragB(vT, LSEQ, dbase,      kb);
    const v16h bv1 = load_fragB(vT, LSEQ, dbase + 16, kb);
    acc0 = __builtin_amdgcn_wmma_f32_16x16x32_f16(false, ap, false, bv0,
                                                  (short)0, acc0, false, false);
    acc1 = __builtin_amdgcn_wmma_f32_16x16x32_f16(false, ap, false, bv1,
                                                  (short)0, acc1, false, false);
    asm volatile("" ::: "memory");   // keep next iter's LDS stores after loads
  }

  // ---- normalize and emit o[q, h, d] as f16 for the output projection ----
#pragma unroll
  for (int r = 0; r < 8; ++r) {
    const float il = 1.0f / li[r];
    const int m = qbase + r + 8 * half;
    oh[(size_t)m * EDIM + dbase + n]      = (_Float16)(acc0[r] * il);
    oh[(size_t)m * EDIM + dbase + 16 + n] = (_Float16)(acc1[r] * il);
  }
}

// ---------------------------------------------------------------------------
extern "C" void kernel_launch(void* const* d_in, const int* in_sizes, int n_in,
                              void* d_out, int out_size, void* d_ws, size_t ws_size,
                              hipStream_t stream) {
  const float* V   = (const float*)d_in[0];
  const float* K   = (const float*)d_in[1];
  const float* Q   = (const float*)d_in[2];
  const float* pos = (const float*)d_in[3];
  const float* Wq  = (const float*)d_in[4];
  const float* Wk  = (const float*)d_in[5];
  const float* Wv  = (const float*)d_in[6];
  const float* Wr  = (const float*)d_in[7];
  const float* Wo  = (const float*)d_in[8];
  const float* bo  = (const float*)d_in[9];
  float* out = (float*)d_out;

  _Float16* Vh  = (_Float16*)d_ws;       // L*E
  _Float16* Kh  = Vh + LE;
  _Float16* Qh  = Kh + LE;
  _Float16* Wqh = Qh + LE;               // E*E
  _Float16* Wkh = Wqh + EE;
  _Float16* Wvh = Wkh + EE;
  _Float16* Woh = Wvh + EE;
  _Float16* qh  = Woh + EE;              // L*E projected q
  _Float16* kh  = qh + LE;               // projected k
  _Float16* vT  = kh + LE;               // projected v, transposed (E x L)
  _Float16* oh  = vT + LE;               // attention output
  float* coeff  = (float*)(oh + LE);     // H * L

  convert_kernel<<<LE / 256, 256, 0, stream>>>(V, K, Q, Wq, Wk, Wv, Wo,
                                               Vh, Kh, Qh, Wqh, Wkh, Wvh, Woh);

  // projections: (L x E) = (L x E) @ W^T ;  1024 waves -> 256 blocks of 128
  wmma_gemm<<<256, 128, 0, stream>>>(Qh, Wqh, LSEQ, EDIM, EDIM,
                                     qh, nullptr, nullptr, nullptr);
  wmma_gemm<<<256, 128, 0, stream>>>(Kh, Wkh, LSEQ, EDIM, EDIM,
                                     kh, nullptr, nullptr, nullptr);
  wmma_gemm<<<256, 128, 0, stream>>>(Vh, Wvh, LSEQ, EDIM, EDIM,
                                     nullptr, vT, nullptr, nullptr);

  coeff_kernel<<<(NH * LSEQ) / 256, 256, 0, stream>>>(qh, Wr, coeff);

  attn_kernel<<<(NH * (LSEQ / 16)) / 4, 128, 0, stream>>>(qh, kh, vT, pos,
                                                          coeff, oh);

  // output projection with bias, f32 result
  wmma_gemm<<<256, 128, 0, stream>>>(oh, Woh, LSEQ, EDIM, EDIM,
                                     nullptr, nullptr, out, bo);
}
